// MultiHeadAttention_5368709120830
// MI455X (gfx1250) — matrix operand order repaired and checked by reading, not compiled, not run on hardware
//
#include <hip/hip_runtime.h>
#include <hip/hip_bf16.h>

#define B_  2
#define T_  2048
#define D_  1024
#define H_  16
#define HD_ 64
#define BT_ (B_*T_)

typedef __attribute__((ext_vector_type(16))) _Float16 v16h;
typedef __attribute__((ext_vector_type(8)))  _Float16 v8h;
typedef __attribute__((ext_vector_type(8)))  float    v8f;

static __device__ __forceinline__ v16h cat8(v8h lo, v8h hi) {
  return __builtin_shufflevector(lo, hi, 0,1,2,3,4,5,6,7,8,9,10,11,12,13,14,15);
}
static __device__ __forceinline__ v8h ld8(const _Float16* p) {
  return *reinterpret_cast<const v8h*>(p);
}

// ---------- CDNA5 async global->LDS copy (ASYNCcnt), with fallback ----------
#if defined(__has_builtin)
#if __has_builtin(__builtin_amdgcn_global_load_async_to_lds_b128)
#define HAVE_ASYNC_LDS 1
#endif
#endif
#ifndef HAVE_ASYNC_LDS
#define HAVE_ASYNC_LDS 0
#endif

typedef int v4i_ __attribute__((vector_size(16)));
typedef __attribute__((address_space(1))) v4i_ gv4i;   // global/device AS
typedef __attribute__((address_space(3))) v4i_ lv4i;   // LDS AS

static __device__ __forceinline__ void copy16_to_lds(const _Float16* gsrc, _Float16* ldst) {
#if HAVE_ASYNC_LDS
  __builtin_amdgcn_global_load_async_to_lds_b128(
      (gv4i*)(void*)gsrc, (lv4i*)(void*)ldst, 0, 0);
#else
  *reinterpret_cast<v8h*>(ldst) = *reinterpret_cast<const v8h*>(gsrc);
#endif
}
static __device__ __forceinline__ void stage_wait() {
#if HAVE_ASYNC_LDS
#if __has_builtin(__builtin_amdgcn_s_wait_asynccnt)
  __builtin_amdgcn_s_wait_asynccnt(0);
#else
  asm volatile("s_wait_asynccnt 0x0" ::: "memory");
#endif
#endif
}

// ---------------- elementwise: f32 -> f16 convert ----------------
__global__ void f32_to_f16_kernel(const float* __restrict__ in,
                                  _Float16* __restrict__ out, int n) {
  int i = blockIdx.x * blockDim.x + threadIdx.x;
  if (i < n) out[i] = (_Float16)in[i];
}

// ---------------- WMMA GEMM: C[M,N] = X[M,K] * W[N,K]^T ----------------
// One wave computes a 32x64 C tile (2 M-subtiles x 4 N-subtiles), K-step 32,
// register double-buffered so loads for step k+1 overlap WMMAs of step k.
// MODE 0: f16 out      MODE 1: f16 out + fused RoPE (N-strip == one head)
// MODE 2: f16 out transposed into Vt[B,H,HD,T]      MODE 3: f32 out
template<int MODE>
__global__ __launch_bounds__(256)
void gemm_kernel(const _Float16* __restrict__ X, const _Float16* __restrict__ W,
                 _Float16* __restrict__ outh, float* __restrict__ outf,
                 const float* __restrict__ cosp, const float* __restrict__ sinp) {
  constexpr int K = D_, N = D_;
  const int lane = threadIdx.x & 31;
  const int wave = threadIdx.x >> 5;
  const int half = lane >> 4;
  const int l16  = lane & 15;
  const int wid  = blockIdx.x * 8 + wave;
  const int mt   = wid >> 4;      // 0..BT/32-1
  const int nt   = wid & 15;      // 0..N/64-1

  const _Float16* xrow0 = X + (size_t)(mt*32 + l16) * K;
  const _Float16* xrow1 = xrow0 + (size_t)16 * K;
  const _Float16* wbase = W + (size_t)(nt*64 + l16) * K + half*8;

  // cheap CDNA5 prefetch of the weight strip (global_prefetch_b8)
#pragma unroll
  for (int s = 0; s < 4; ++s) __builtin_prefetch(wbase + (size_t)s*16*K, 0, 1);

  v8f zero = {};
  v8f acc[2][4];
#pragma unroll
  for (int mi = 0; mi < 2; ++mi)
#pragma unroll
    for (int s = 0; s < 4; ++s) acc[mi][s] = zero;

  auto loadA = [&](int k0, v16h& a0, v16h& a1) {
    a0 = cat8(ld8(xrow0 + k0 + half*8), ld8(xrow0 + k0 + 16 + half*8));
    a1 = cat8(ld8(xrow1 + k0 + half*8), ld8(xrow1 + k0 + 16 + half*8));
  };
  auto loadB = [&](int k0, v16h* bb) {
#pragma unroll
    for (int s = 0; s < 4; ++s) {
      const _Float16* wrow = wbase + (size_t)s*16*K + k0;
      bb[s] = cat8(ld8(wrow), ld8(wrow + 16));
    }
  };
  auto mmac = [&](const v16h& a0, const v16h& a1, const v16h* bb) {
#pragma unroll
    for (int s = 0; s < 4; ++s) {
      acc[0][s] = __builtin_amdgcn_wmma_f32_16x16x32_f16(
          false, a0, false, bb[s], (short)0, acc[0][s], false, false);
      acc[1][s] = __builtin_amdgcn_wmma_f32_16x16x32_f16(
          false, a1, false, bb[s], (short)0, acc[1][s], false, false);
    }
  };

  v16h a0, a1, bb[4];
  loadA(0, a0, a1);
  loadB(0, bb);
  for (int k0 = 0; k0 < K - 32; k0 += 32) {
    v16h na0, na1, nb[4];
    loadA(k0 + 32, na0, na1);      // prefetch next step into registers
    loadB(k0 + 32, nb);
    mmac(a0, a1, bb);              // compute current step
    a0 = na0; a1 = na1;
#pragma unroll
    for (int s = 0; s < 4; ++s) bb[s] = nb[s];
  }
  mmac(a0, a1, bb);                // peeled last step

  // C layout: vgpr r, lane<16 -> (M=r, N=l16); lane>=16 -> (M=r+8, N=l16)
#pragma unroll
  for (int mi = 0; mi < 2; ++mi) {
    if constexpr (MODE == 1) {
      // fused RoPE: hd pair (hd, hd+32) == subtile pair (s, s+2), same lane
#pragma unroll
      for (int s = 0; s < 2; ++s) {
#pragma unroll
        for (int r = 0; r < 8; ++r) {
          const size_t m  = (size_t)(mt*32 + mi*16 + r + half*8);
          const int    hd = s*16 + l16;
          const float c1 = cosp[m*HD_ + hd],      s1 = sinp[m*HD_ + hd];
          const float c2 = cosp[m*HD_ + hd + 32], s2 = sinp[m*HD_ + hd + 32];
          const float q1 = acc[mi][s][r], q2 = acc[mi][s+2][r];
          outh[m*N + nt*64 + hd]      = (_Float16)(q1*c1 - q2*s1);
          outh[m*N + nt*64 + hd + 32] = (_Float16)(q2*c2 + q1*s2);
        }
      }
    } else {
#pragma unroll
      for (int s = 0; s < 4; ++s) {
#pragma unroll
        for (int r = 0; r < 8; ++r) {
          const size_t m = (size_t)(mt*32 + mi*16 + r + half*8);
          const int    n = nt*64 + s*16 + l16;
          if constexpr (MODE == 0) {
            outh[m*N + n] = (_Float16)acc[mi][s][r];
          } else if constexpr (MODE == 2) {
            // store straight into Vt[B,H,HD,T]; head h == nt, hd == s*16+l16
            const size_t b = m >> 11, t = m & (T_ - 1);
            outh[((b*H_ + nt)*HD_ + (size_t)(s*16 + l16))*T_ + t] =
                (_Float16)acc[mi][s][r];
          } else {
            outf[m*N + n] = acc[mi][s][r];
          }
        }
      }
    }
  }
}

// ---------------- causal flash attention ----------------
// One block = 8 waves = 8 consecutive 16-query tiles of one (b,h).
// K (32x64) and V (64x32) tiles staged ONCE per block into LDS via async
// global->LDS DMA (double-buffered: DMA of block kb+1 overlaps softmax of kb).
__global__ __launch_bounds__(256)
void flash_attn_kernel(const _Float16* __restrict__ Q,
                       const _Float16* __restrict__ Kb,
                       const _Float16* __restrict__ Vt,
                       _Float16* __restrict__ O) {
  __shared__ _Float16 ldsK[2][32 * 64];   // [key_local][hd]
  __shared__ _Float16 ldsV[2][64 * 32];   // [hd][key_local]
  __shared__ _Float16 pbuf[8][16 * 32];   // per-wave P staging

  const int lane = threadIdx.x & 31;
  const int wave = threadIdx.x >> 5;
  const int half = lane >> 4;
  const int l16  = lane & 15;

  const int nblk = T_ / 128;              // 16 q-blocks per (b,h)
  const int qblk = blockIdx.x % nblk;
  const int bh   = blockIdx.x / nblk;
  const int h    = bh % H_;
  const int b    = bh / H_;
  const int qt   = qblk*8 + wave;
  const int qt0  = qt * 16;
  const size_t rowbase = (size_t)b * T_;
  const _Float16* vtbase = Vt + (size_t)bh * HD_ * T_;

  // Q A-fragments (head dim 64 -> two K=32 fragments), loaded once
  const _Float16* qrow = Q + (rowbase + qt0 + l16) * D_ + h * HD_;
  const v16h aq0 = cat8(ld8(qrow + half*8),      ld8(qrow + 16 + half*8));
  const v16h aq1 = cat8(ld8(qrow + 32 + half*8), ld8(qrow + 48 + half*8));

  v8f zero = {};
  v8f o_acc[4];
#pragma unroll
  for (int s = 0; s < 4; ++s) o_acc[s] = zero;
  float mrow[8], lrow[8];
#pragma unroll
  for (int r = 0; r < 8; ++r) { mrow[r] = -1e30f; lrow[r] = 0.0f; }

  const float scale = 0.125f;                       // 1/sqrt(64)
  const int nkb_wave  = qt/2 + 1;                   // this wave's causal extent
  const int nkb_block = (qblk*8 + 7)/2 + 1;         // block's max extent

  // producers: wave 0 stages K tile, wave 1 stages V tile (wave-uniform)
  auto stage = [&](int kb, int buf) {
    const int key0 = kb * 32;
    if (wave == 0) {
      const int kl = lane >> 3;          // 4 keys per issue
      const int co = (lane & 7) * 8;     // 8 lanes cover 64 halves/row
#pragma unroll
      for (int i = 0; i < 8; ++i) {
        const int key_local = i*4 + kl;
        copy16_to_lds(Kb + (rowbase + key0 + key_local)*D_ + h*HD_ + co,
                      &ldsK[buf][key_local*64 + co]);
      }
    } else if (wave == 1) {
      const int vl = lane >> 2;          // 8 hd rows per issue
      const int co = (lane & 3) * 8;     // 4 lanes cover 32 halves/row
#pragma unroll
      for (int i = 0; i < 8; ++i) {
        const int hd = i*8 + vl;
        copy16_to_lds(vtbase + (size_t)hd*T_ + key0 + co,
                      &ldsV[buf][hd*32 + co]);
      }
    }
  };

  stage(0, 0);
  stage_wait();
  __syncthreads();

  for (int kb = 0; kb < nkb_block; ++kb) {
    const int buf = kb & 1;
    if (kb + 1 < nkb_block) stage(kb + 1, buf ^ 1);   // overlap DMA w/ compute

    if (kb < nkb_wave) {                              // wave-uniform gate
      const int key0 = kb * 32;

      // --- S = Q K^T for two 16-key sub-tiles, K fragments from LDS ---
      v8f s_c[2];
#pragma unroll
      for (int c = 0; c < 2; ++c) {
        const _Float16* krow = &ldsK[buf][(c*16 + l16)*64 + half*8];
        v16h bk0 = cat8(ld8(krow),      ld8(krow + 16));
        v16h bk1 = cat8(ld8(krow + 32), ld8(krow + 48));
        v8f s = zero;
        s = __builtin_amdgcn_wmma_f32_16x16x32_f16(false, aq0, false, bk0,
                                                   (short)0, s, false, false);
        s = __builtin_amdgcn_wmma_f32_16x16x32_f16(false, aq1, false, bk1,
                                                   (short)0, s, false, false);
        s_c[c] = s;
      }

      // --- mask + online softmax (rows live across 16-lane groups) ---
      float alpha_r[8];
#pragma unroll
      for (int r = 0; r < 8; ++r) {
        const int qidx = qt0 + r + half*8;
        float v0 = s_c[0][r] * scale;
        float v1 = s_c[1][r] * scale;
        if (key0 + l16 > qidx)      v0 = -1e30f;
        if (key0 + 16 + l16 > qidx) v1 = -1e30f;

        float mx = fmaxf(v0, v1);
        mx = fmaxf(mx, __shfl_xor(mx, 1, 32));
        mx = fmaxf(mx, __shfl_xor(mx, 2, 32));
        mx = fmaxf(mx, __shfl_xor(mx, 4, 32));
        mx = fmaxf(mx, __shfl_xor(mx, 8, 32));
        const float mnew = fmaxf(mrow[r], mx);

        const float alpha = __expf(mrow[r] - mnew);
        const float e0 = __expf(v0 - mnew);
        const float e1 = __expf(v1 - mnew);
        float rs = e0 + e1;
        rs += __shfl_xor(rs, 1, 32);
        rs += __shfl_xor(rs, 2, 32);
        rs += __shfl_xor(rs, 4, 32);
        rs += __shfl_xor(rs, 8, 32);

        lrow[r] = lrow[r] * alpha + rs;
        mrow[r] = mnew;
        alpha_r[r] = alpha;

        pbuf[wave][(r + half*8)*32 + l16]      = (_Float16)e0;
        pbuf[wave][(r + half*8)*32 + 16 + l16] = (_Float16)e1;
      }

#pragma unroll
      for (int s = 0; s < 4; ++s)
#pragma unroll
        for (int r = 0; r < 8; ++r) o_acc[s][r] *= alpha_r[r];

      // --- P back as A-fragment; O += P * V (V fragments from LDS) ---
      const _Float16* pr = &pbuf[wave][l16 * 32];
      v16h ap = cat8(ld8(pr + half*8), ld8(pr + 16 + half*8));
#pragma unroll
      for (int s = 0; s < 4; ++s) {
        const _Float16* vrow = &ldsV[buf][(s*16 + l16)*32 + half*8];
        v16h bv = cat8(ld8(vrow), ld8(vrow + 16));
        o_acc[s] = __builtin_amdgcn_wmma_f32_16x16x32_f16(
            false, ap, false, bv, (short)0, o_acc[s], false, false);
      }
    }

    stage_wait();        // producers drain ASYNCcnt before handoff
    __syncthreads();
  }

  // --- finalize: divide by row sums, store f16 ---
#pragma unroll
  for (int s = 0; s < 4; ++s)
#pragma unroll
    for (int r = 0; r < 8; ++r) {
      const size_t m = rowbase + qt0 + r + half*8;
      O[m * D_ + h * HD_ + s*16 + l16] = (_Float16)(o_acc[s][r] / lrow[r]);
    }
}

// ---------------- host-side orchestration ----------------
extern "C" void kernel_launch(void* const* d_in, const int* in_sizes, int n_in,
                              void* d_out, int out_size, void* d_ws, size_t ws_size,
                              hipStream_t stream) {
  const float* x    = (const float*)d_in[0];
  const float* cosp = (const float*)d_in[1];
  const float* sinp = (const float*)d_in[2];
  const float* Wq   = (const float*)d_in[3];
  const float* Wk   = (const float*)d_in[4];
  const float* Wv   = (const float*)d_in[5];
  const float* Wp   = (const float*)d_in[6];
  float* out = (float*)d_out;

  char* ws = (char*)d_ws;
  size_t off = 0;
  const size_t xhN = (size_t)BT_ * D_;
  const size_t wN  = (size_t)D_ * D_;
  _Float16* Xh  = (_Float16*)(ws + off); off += xhN * 2;
  _Float16* Wqh = (_Float16*)(ws + off); off += wN * 2;
  _Float16* Wkh = (_Float16*)(ws + off); off += wN * 2;
  _Float16* Wvh = (_Float16*)(ws + off); off += wN * 2;
  _Float16* Wph = (_Float16*)(ws + off); off += wN * 2;
  _Float16* Qh  = (_Float16*)(ws + off); off += xhN * 2;
  _Float16* Kh  = (_Float16*)(ws + off); off += xhN * 2;
  _Float16* Vth = (_Float16*)(ws + off); off += xhN * 2;
  _Float16* Oh  = (_Float16*)(ws + off); off += xhN * 2;

  // 1) f32 -> f16 converts
  f32_to_f16_kernel<<<(int)((xhN + 255) / 256), 256, 0, stream>>>(x,  Xh,  (int)xhN);
  f32_to_f16_kernel<<<(int)((wN  + 255) / 256), 256, 0, stream>>>(Wq, Wqh, (int)wN);
  f32_to_f16_kernel<<<(int)((wN  + 255) / 256), 256, 0, stream>>>(Wk, Wkh, (int)wN);
  f32_to_f16_kernel<<<(int)((wN  + 255) / 256), 256, 0, stream>>>(Wv, Wvh, (int)wN);
  f32_to_f16_kernel<<<(int)((wN  + 255) / 256), 256, 0, stream>>>(Wp, Wph, (int)wN);

  // 2) projections: (BT/32)*(D/64) wave-tiles / 8 waves per block = 256 blocks
  const int gblocks = (BT_ / 32) * (D_ / 64) / 8;
  // Q,K with fused RoPE epilogue
  gemm_kernel<1><<<gblocks, 256, 0, stream>>>(Xh, Wqh, Qh, nullptr, cosp, sinp);
  gemm_kernel<1><<<gblocks, 256, 0, stream>>>(Xh, Wkh, Kh, nullptr, cosp, sinp);
  // V with fused transpose-to-Vt epilogue
  gemm_kernel<2><<<gblocks, 256, 0, stream>>>(Xh, Wvh, Vth, nullptr, nullptr, nullptr);

  // 3) causal flash attention: one block per (b,h,128-query slab)
  const int fblocks = B_ * H_ * (T_ / 128);   // 512
  flash_attn_kernel<<<fblocks, 256, 0, stream>>>(Qh, Kh, Vth, Oh);

  // 4) output projection -> f32 d_out
  gemm_kernel<3><<<gblocks, 256, 0, stream>>>(Oh, Wph, nullptr, out, nullptr, nullptr);
}
